// RelPositionMultiHeadAttention_39316130627951
// MI455X (gfx1250) — compile-verified, matrix operand order checked by reading, not run on hardware
//
#include <hip/hip_runtime.h>
#include <hip/hip_bf16.h>

// ---------------------------------------------------------------------------
// RelPositionMultiHeadAttention for MI455X (gfx1250), wave32 + WMMA bf16.
// B=8, S=1024, D=512, H=8, DK=64, P=2S-1=2047.
// All matmuls use v_wmma_f32_16x16x32_bf16 (fp32 accum). fp32 operands are
// converted to bf16 ONCE in a separate bandwidth-bound pass, so GEMM inner
// loops are pure vector-load + v_wmma (+ global_prefetch for the A stream).
// ---------------------------------------------------------------------------

#define KB 8        // batch
#define KS 1024     // sequence
#define KD 512      // model dim
#define KH 8        // heads
#define KDK 64      // head dim
#define KP 2047     // 2*S-1

typedef __attribute__((ext_vector_type(16))) __bf16 v16bf;
typedef __attribute__((ext_vector_type(8)))  float  v8f;

__device__ __forceinline__ __bf16 f2bf(float f) {
  unsigned int u = __builtin_bit_cast(unsigned int, f);
  unsigned int r = u + 0x7FFFu + ((u >> 16) & 1u);   // round-to-nearest-even
  unsigned short h = (unsigned short)(r >> 16);
  return __builtin_bit_cast(__bf16, h);
}

__device__ __forceinline__ unsigned int pk2bf(float a, float b) {
  unsigned int ua = __builtin_bit_cast(unsigned int, a);
  unsigned int ub = __builtin_bit_cast(unsigned int, b);
  ua = ua + 0x7FFFu + ((ua >> 16) & 1u);
  ub = ub + 0x7FFFu + ((ub >> 16) & 1u);
  return (ua >> 16) | (ub & 0xFFFF0000u);
}

__device__ __forceinline__ v16bf load_frag(const __bf16* __restrict__ p) {
  return *reinterpret_cast<const v16bf*>(p);  // 32B aligned by construction
}

__device__ __forceinline__ v8f wmma_bf16(v16bf a, v16bf b, v8f c) {
  return __builtin_amdgcn_wmma_f32_16x16x32_bf16(false, a, false, b,
                                                 (short)0, c, false, false);
}

// ---------------------------------------------------------------------------
// One-shot fp32 -> bf16 conversion (bandwidth bound, done once per tensor).
// ---------------------------------------------------------------------------
__global__ __launch_bounds__(256) void cvt_bf16_kernel(
    const float4* __restrict__ in, uint2* __restrict__ out, int n4) {
  int i = blockIdx.x * 256 + threadIdx.x;
  if (i < n4) {
    float4 f = in[i];
    out[i] = make_uint2(pk2bf(f.x, f.y), pk2bf(f.z, f.w));
  }
}

// ---------------------------------------------------------------------------
// Pass A: C = A(bf16, Nrows x K) @ W(bf16, M x K)^T, dual biased bf16 outputs.
// out1 += bias1+bias2 ; out2 (if non-null) += bias1+bias3.
// mode 0: head-major (B,H,S,DK);  mode 1: v transposed (B,H,DK,S);
// mode 2: positional head-major (H,P,DK).
// Block = 256 threads = 8 waves; wave w -> 16 rows x 64 cols (4 acc tiles).
// ---------------------------------------------------------------------------
__global__ __launch_bounds__(256) void proj_gemm_kernel(
    const __bf16* __restrict__ A, const __bf16* __restrict__ W,
    const float* __restrict__ bias1, const float* __restrict__ bias2,
    const float* __restrict__ bias3,
    __bf16* __restrict__ out1, __bf16* __restrict__ out2,
    int Nrows, int K, int M, int mode) {
  const int lane = threadIdx.x & 31;
  const int wave = threadIdx.x >> 5;
  const int row0 = blockIdx.y * 128 + wave * 16;
  const int col0 = blockIdx.x * 64;
  const int fr   = lane & 15;           // fragment row/col within tile
  const int kh   = (lane >> 4) * 16;    // K-half offset

  int arow = row0 + fr;
  if (arow >= Nrows) arow = Nrows - 1;  // clamp (P=2047 tail)
  const __bf16* ap = A + (size_t)arow * K + kh;
  const __bf16* wp = W + (size_t)(col0 + fr) * K + kh;

  v8f acc[4] = {v8f{}, v8f{}, v8f{}, v8f{}};
  for (int k0 = 0; k0 < K; k0 += 32) {
    if (k0 + 32 < K) __builtin_prefetch(ap + k0 + 32, 0, 1);  // global_prefetch
    v16bf afrag = load_frag(ap + k0);
#pragma unroll
    for (int j = 0; j < 4; ++j) {
      v16bf bfrag = load_frag(wp + (size_t)j * 16 * K + k0);
      acc[j] = wmma_bf16(afrag, bfrag, acc[j]);
    }
  }

  const int ocol   = lane & 15;
  const int rowadd = (lane >> 4) * 8;   // lanes 16..31 hold rows v+8
#pragma unroll
  for (int j = 0; j < 4; ++j) {
#pragma unroll
    for (int v = 0; v < 8; ++v) {
      int rr = row0 + v + rowadd;
      int cc = col0 + j * 16 + ocol;
      if (rr < Nrows) {
        float val = acc[j][v];
        if (bias1) val += bias1[cc];
        int bb = rr >> 10, ss = rr & (KS - 1);
        int hh = cc >> 6,  dk = cc & (KDK - 1);
        size_t idx;
        if (mode == 0)      idx = ((size_t)((bb * KH + hh) * KS + ss)) * KDK + dk;
        else if (mode == 1) idx = ((size_t)((bb * KH + hh) * KDK + dk)) * KS + ss;
        else                idx = ((size_t)(hh * KP + rr)) * KDK + dk;
        float v1 = bias2 ? val + bias2[cc] : val;
        out1[idx] = f2bf(v1);
        if (out2) {
          float v2 = bias3 ? val + bias3[cc] : val;
          out2[idx] = f2bf(v2);
        }
      }
    }
  }
}

// ---------------------------------------------------------------------------
// BD raw GEMM (one batch b): bd[h,s,p] = qv[b,h,s,:] . pos[h,p,:]   (K = 64)
// ---------------------------------------------------------------------------
__global__ __launch_bounds__(256) void bd_gemm_kernel(
    const __bf16* __restrict__ qv, const __bf16* __restrict__ pos,
    float* __restrict__ bd, int b) {
  const int lane = threadIdx.x & 31;
  const int wave = threadIdx.x >> 5;
  const int h    = blockIdx.z;
  const int row0 = blockIdx.y * 128 + wave * 16;
  const int col0 = blockIdx.x * 64;
  const int fr   = lane & 15;
  const int kh   = (lane >> 4) * 16;

  const __bf16* Ah = qv  + ((size_t)(b * KH + h) * KS) * KDK;
  const __bf16* Bh = pos + (size_t)h * KP * KDK;

  v8f acc[4] = {v8f{}, v8f{}, v8f{}, v8f{}};
#pragma unroll
  for (int k0 = 0; k0 < KDK; k0 += 32) {
    v16bf afrag = load_frag(Ah + (size_t)(row0 + fr) * KDK + k0 + kh);
#pragma unroll
    for (int j = 0; j < 4; ++j) {
      int bcol = col0 + j * 16 + fr;
      if (bcol >= KP) bcol = KP - 1;
      v16bf bfrag = load_frag(Bh + (size_t)bcol * KDK + k0 + kh);
      acc[j] = wmma_bf16(afrag, bfrag, acc[j]);
    }
  }

  const int ocol = lane & 15, rowadd = (lane >> 4) * 8;
#pragma unroll
  for (int j = 0; j < 4; ++j)
#pragma unroll
    for (int v = 0; v < 8; ++v) {
      int rr = row0 + v + rowadd, cc = col0 + j * 16 + ocol;
      if (cc < KP)
        bd[((size_t)(h * KS + rr)) * KP + cc] = acc[j][v];
    }
}

// ---------------------------------------------------------------------------
// Scores (one batch b): AC = qu . kh^T via WMMA (K=64), plus rel-shifted BD
// gathered with the exact flat-reshape index  bd_flat[(h*S+q)*P + (S-q+t)],
// scale 1/sqrt(64), key mask -> -10000.
// ---------------------------------------------------------------------------
__global__ __launch_bounds__(256) void scores_kernel(
    const __bf16* __restrict__ qu, const __bf16* __restrict__ khm,
    const float* __restrict__ bd, const unsigned char* __restrict__ maskb,
    float* __restrict__ scores, int b) {
  const int lane = threadIdx.x & 31;
  const int wave = threadIdx.x >> 5;
  const int h    = blockIdx.z;
  const int row0 = blockIdx.y * 128 + wave * 16;
  const int col0 = blockIdx.x * 64;
  const int fr   = lane & 15;
  const int kh   = (lane >> 4) * 16;

  const __bf16* Ah = qu  + ((size_t)(b * KH + h) * KS) * KDK;
  const __bf16* Bh = khm + ((size_t)(b * KH + h) * KS) * KDK;

  v8f acc[4] = {v8f{}, v8f{}, v8f{}, v8f{}};
#pragma unroll
  for (int k0 = 0; k0 < KDK; k0 += 32) {
    v16bf afrag = load_frag(Ah + (size_t)(row0 + fr) * KDK + k0 + kh);
#pragma unroll
    for (int j = 0; j < 4; ++j) {
      v16bf bfrag = load_frag(Bh + (size_t)(col0 + j * 16 + fr) * KDK + k0 + kh);
      acc[j] = wmma_bf16(afrag, bfrag, acc[j]);
    }
  }

  const int ocol = lane & 15, rowadd = (lane >> 4) * 8;
#pragma unroll
  for (int j = 0; j < 4; ++j)
#pragma unroll
    for (int v = 0; v < 8; ++v) {
      int q = row0 + v + rowadd;
      int t = col0 + j * 16 + ocol;
      size_t fl = ((size_t)(h * KS + q)) * KP + (size_t)(KS - q + t);
      float val = (acc[j][v] + bd[fl]) * 0.125f;   // 1/sqrt(64)
      if (maskb[t]) val = -10000.0f;
      scores[((size_t)(h * KS + q)) * KS + t] = val;
    }
}

// ---------------------------------------------------------------------------
// Row softmax (one batch b): 1024-wide row per block, bf16 output.
// ---------------------------------------------------------------------------
__global__ __launch_bounds__(256) void softmax_kernel(
    const float* __restrict__ scores, __bf16* __restrict__ attn) {
  const int q = blockIdx.x, h = blockIdx.y, tid = threadIdx.x;
  const float* row = scores + ((size_t)(h * KS + q)) * KS;
  __bf16* orow     = attn   + ((size_t)(h * KS + q)) * KS;
  __shared__ float red[256];

  float vals[4];
  float m = -3.4e38f;
#pragma unroll
  for (int i = 0; i < 4; ++i) { vals[i] = row[tid + i * 256]; m = fmaxf(m, vals[i]); }
  red[tid] = m; __syncthreads();
  for (int s = 128; s > 0; s >>= 1) {
    if (tid < s) red[tid] = fmaxf(red[tid], red[tid + s]);
    __syncthreads();
  }
  m = red[0]; __syncthreads();

  float sum = 0.f;
#pragma unroll
  for (int i = 0; i < 4; ++i) { vals[i] = __expf(vals[i] - m); sum += vals[i]; }
  red[tid] = sum; __syncthreads();
  for (int s = 128; s > 0; s >>= 1) {
    if (tid < s) red[tid] += red[tid + s];
    __syncthreads();
  }
  float inv = 1.f / red[0];
#pragma unroll
  for (int i = 0; i < 4; ++i) orow[tid + i * 256] = f2bf(vals[i] * inv);
}

// ---------------------------------------------------------------------------
// attn @ v (one batch b): x[s,dk] = sum_t attn[h,s,t] * vt[b,h,dk,t], K=1024.
// Scatter result into token-major xcat (B,S,D) for the output projection.
// ---------------------------------------------------------------------------
__global__ __launch_bounds__(256) void av_gemm_kernel(
    const __bf16* __restrict__ attn, const __bf16* __restrict__ vt,
    __bf16* __restrict__ xcat, int b) {
  const int lane = threadIdx.x & 31;
  const int wave = threadIdx.x >> 5;
  const int h    = blockIdx.z;
  const int row0 = blockIdx.y * 128 + wave * 16;
  const int fr   = lane & 15;
  const int kh   = (lane >> 4) * 16;

  const __bf16* Ah = attn + (size_t)h * KS * KS + (size_t)(row0 + fr) * KS + kh;
  const __bf16* Bh = vt   + ((size_t)(b * KH + h) * KDK) * KS + (size_t)fr * KS + kh;

  v8f acc[4] = {v8f{}, v8f{}, v8f{}, v8f{}};
  for (int k0 = 0; k0 < KS; k0 += 32) {
    if (k0 + 32 < KS) __builtin_prefetch(Ah + k0 + 32, 0, 1);  // global_prefetch
    v16bf afrag = load_frag(Ah + k0);
#pragma unroll
    for (int j = 0; j < 4; ++j) {  // j*16+fr = dk
      v16bf bfrag = load_frag(Bh + (size_t)j * 16 * KS + k0);
      acc[j] = wmma_bf16(afrag, bfrag, acc[j]);
    }
  }

  const int ocol = lane & 15, rowadd = (lane >> 4) * 8;
#pragma unroll
  for (int j = 0; j < 4; ++j)
#pragma unroll
    for (int v = 0; v < 8; ++v) {
      int ss = row0 + v + rowadd;
      int dk = j * 16 + ocol;
      xcat[((size_t)(b * KS + ss)) * KD + h * KDK + dk] = f2bf(acc[j][v]);
    }
}

// ---------------------------------------------------------------------------
// Output projection: out = xcat(bf16, 8192x512) @ Wo(bf16,512x512)^T + bo.
// ---------------------------------------------------------------------------
__global__ __launch_bounds__(256) void out_gemm_kernel(
    const __bf16* __restrict__ A, const __bf16* __restrict__ W,
    const float* __restrict__ bias, float* __restrict__ out) {
  const int lane = threadIdx.x & 31;
  const int wave = threadIdx.x >> 5;
  const int row0 = blockIdx.y * 128 + wave * 16;
  const int col0 = blockIdx.x * 64;
  const int fr   = lane & 15;
  const int kh   = (lane >> 4) * 16;

  const __bf16* ap = A + (size_t)(row0 + fr) * KD + kh;
  const __bf16* wp = W + (size_t)(col0 + fr) * KD + kh;

  v8f acc[4] = {v8f{}, v8f{}, v8f{}, v8f{}};
  for (int k0 = 0; k0 < KD; k0 += 32) {
    if (k0 + 32 < KD) __builtin_prefetch(ap + k0 + 32, 0, 1);  // global_prefetch
    v16bf afrag = load_frag(ap + k0);
#pragma unroll
    for (int j = 0; j < 4; ++j) {
      v16bf bfrag = load_frag(wp + (size_t)j * 16 * KD + k0);
      acc[j] = wmma_bf16(afrag, bfrag, acc[j]);
    }
  }

  const int ocol = lane & 15, rowadd = (lane >> 4) * 8;
#pragma unroll
  for (int j = 0; j < 4; ++j)
#pragma unroll
    for (int v = 0; v < 8; ++v) {
      int rr = row0 + v + rowadd;
      int cc = col0 + j * 16 + ocol;
      out[(size_t)rr * KD + cc] = acc[j][v] + bias[cc];
    }
}

// ---------------------------------------------------------------------------
// Host orchestration.
// ---------------------------------------------------------------------------
extern "C" void kernel_launch(void* const* d_in, const int* in_sizes, int n_in,
                              void* d_out, int out_size, void* d_ws, size_t ws_size,
                              hipStream_t stream) {
  const float* q   = (const float*)d_in[0];
  const float* k   = (const float*)d_in[1];
  const float* v   = (const float*)d_in[2];
  const float* pe  = (const float*)d_in[3];
  const unsigned char* mask = (const unsigned char*)d_in[4];
  const float* Wq  = (const float*)d_in[5];
  const float* bq  = (const float*)d_in[6];
  const float* Wk  = (const float*)d_in[7];
  const float* bk  = (const float*)d_in[8];
  const float* Wv  = (const float*)d_in[9];
  const float* bv  = (const float*)d_in[10];
  const float* Wo  = (const float*)d_in[11];
  const float* bo  = (const float*)d_in[12];
  const float* Wp  = (const float*)d_in[13];
  const float* pbu = (const float*)d_in[14];  // (H,DK) flat == D
  const float* pbv = (const float*)d_in[15];

  // Workspace carve (bytes, all 256-aligned).
  char* ws = (char*)d_ws;
  const size_t projB = (size_t)KB * KH * KS * KDK * 2;        // 8 MiB each
  const size_t posB  = (size_t)KH * KP * KDK * 2;
  const size_t wB    = (size_t)KD * KD * 2;
  size_t off = 0;
  __bf16* qu   = (__bf16*)(ws + off); off += projB;
  __bf16* qv   = (__bf16*)(ws + off); off += projB;
  __bf16* khm  = (__bf16*)(ws + off); off += projB;
  __bf16* vt   = (__bf16*)(ws + off); off += projB;
  __bf16* pos  = (__bf16*)(ws + off); off += posB;            // (H,P,DK)
  __bf16* xcat = (__bf16*)(ws + off); off += projB;
  float*  bd   = (float*) (ws + off); off += (size_t)KH * KS * KP * 4;
  float*  sc   = (float*) (ws + off); off += (size_t)KH * KS * KS * 4;
  __bf16* attn = (__bf16*)(ws + off); off += (size_t)KH * KS * KS * 2;
  __bf16* qbf  = (__bf16*)(ws + off); off += projB;           // (B,S,D)
  __bf16* kbf  = (__bf16*)(ws + off); off += projB;
  __bf16* vbf  = (__bf16*)(ws + off); off += projB;
  __bf16* pebf = (__bf16*)(ws + off); off += ((size_t)KP * KD * 2 + 255) & ~(size_t)255;
  __bf16* wqb  = (__bf16*)(ws + off); off += wB;
  __bf16* wkb  = (__bf16*)(ws + off); off += wB;
  __bf16* wvb  = (__bf16*)(ws + off); off += wB;
  __bf16* wob  = (__bf16*)(ws + off); off += wB;
  __bf16* wpb  = (__bf16*)(ws + off); off += wB;

  const int NTOK = KB * KS;  // 8192
  dim3 blk(256);

  // One-shot fp32 -> bf16 conversions (bandwidth bound).
  auto cvt = [&](const float* src, __bf16* dst, size_t n) {
    int n4 = (int)(n / 4);
    cvt_bf16_kernel<<<dim3((n4 + 255) / 256), blk, 0, stream>>>(
        (const float4*)src, (uint2*)dst, n4);
  };
  cvt(q,  qbf,  (size_t)NTOK * KD);
  cvt(k,  kbf,  (size_t)NTOK * KD);
  cvt(v,  vbf,  (size_t)NTOK * KD);
  cvt(pe, pebf, (size_t)KP * KD);
  cvt(Wq, wqb,  (size_t)KD * KD);
  cvt(Wk, wkb,  (size_t)KD * KD);
  cvt(Wv, wvb,  (size_t)KD * KD);
  cvt(Wo, wob,  (size_t)KD * KD);
  cvt(Wp, wpb,  (size_t)KD * KD);

  // Pass A: projection GEMMs (q projection writes qu AND qv in one pass).
  dim3 gP(KD / 64, NTOK / 128);
  proj_gemm_kernel<<<gP, blk, 0, stream>>>(qbf, wqb, bq, pbu, pbv, qu, qv,
                                           NTOK, KD, KD, 0);
  proj_gemm_kernel<<<gP, blk, 0, stream>>>(kbf, wkb, bk, nullptr, nullptr, khm,
                                           nullptr, NTOK, KD, KD, 0);
  proj_gemm_kernel<<<gP, blk, 0, stream>>>(vbf, wvb, bv, nullptr, nullptr, vt,
                                           nullptr, NTOK, KD, KD, 1);
  dim3 gPos(KD / 64, (KP + 127) / 128);
  proj_gemm_kernel<<<gPos, blk, 0, stream>>>(pebf, wpb, nullptr, nullptr, nullptr,
                                             pos, nullptr, KP, KD, KD, 2);

  // Pass B: per-batch attention (stream-serialized, bd/sc/attn reused).
  for (int b = 0; b < KB; ++b) {
    dim3 gBD((KP + 63) / 64, KS / 128, KH);
    bd_gemm_kernel<<<gBD, blk, 0, stream>>>(qv, pos, bd, b);

    dim3 gSC(KS / 64, KS / 128, KH);
    scores_kernel<<<gSC, blk, 0, stream>>>(qu, khm, bd, mask + (size_t)b * KS, sc, b);

    dim3 gSM(KS, KH);
    softmax_kernel<<<gSM, blk, 0, stream>>>(sc, attn);

    dim3 gAV(1, KS / 128, KH);
    av_gemm_kernel<<<gAV, blk, 0, stream>>>(attn, vt, xcat, b);
  }

  // Output projection -> fp32 d_out.
  dim3 gO(KD / 64, NTOK / 128);
  out_gemm_kernel<<<gO, blk, 0, stream>>>(xcat, wob, bo, (float*)d_out);
}
